// HANLayer_60430189854754
// MI455X (gfx1250) — matrix operand order repaired and checked by reading, not compiled, not run on hardware
//
#include <hip/hip_runtime.h>
#include <hip/hip_bf16.h>
#include <math.h>

// ---------------------------------------------------------------------------
// HAN layer (6-relation GATConv, H=4, D=64, IN=128) for gfx1250 / MI455X.
// Dense projections via v_wmma_f32_16x16x32_bf16 (K=128 = 4 x K32 steps).
// Edge softmax: max/sum atomic passes -> per-edge alpha -> fused scatter.
// ---------------------------------------------------------------------------

typedef __bf16 bf16_t;
typedef __attribute__((ext_vector_type(8)))  bf16_t v8bf;
typedef __attribute__((ext_vector_type(16))) bf16_t v16bf;
typedef __attribute__((ext_vector_type(8)))  float  v8f;
typedef __attribute__((ext_vector_type(4)))  float  f32x4;

#define N_USER   50000
#define N_ARTIST 20000
#define N_TAG    5000
#define IN_F     128
#define NHEAD    4
#define DOUT     64
#define HD       256       // NHEAD * DOUT
#define E_MAX    500000
#define NEG_SLOPE 0.2f

static inline int cdiv(long a, int b) { return (int)((a + b - 1) / b); }

// -------------------------------- kernels ----------------------------------

__global__ void cvt_f32_to_bf16(const float* __restrict__ x, bf16_t* __restrict__ y, long n) {
    long i = (long)blockIdx.x * blockDim.x + threadIdx.x;
    if (i < n) y[i] = (bf16_t)x[i];
}

// wl[i][h] = sum_d W[h*64+d][i]*al[h*64+d]  (and same with ar -> wr). 128 thr.
__global__ void fold_attn(const float* __restrict__ W, const float* __restrict__ al,
                          const float* __restrict__ ar, float* __restrict__ wl,
                          float* __restrict__ wr) {
    int i = threadIdx.x;            // 0..127 (input-feature index)
    float sl[NHEAD] = {0.f, 0.f, 0.f, 0.f};
    float sr[NHEAD] = {0.f, 0.f, 0.f, 0.f};
    for (int h = 0; h < NHEAD; ++h)
        for (int d = 0; d < DOUT; ++d) {
            float w = W[(h * DOUT + d) * IN_F + i];
            sl[h] += w * al[h * DOUT + d];
            sr[h] += w * ar[h * DOUT + d];
        }
    for (int h = 0; h < NHEAD; ++h) { wl[i * NHEAD + h] = sl[h]; wr[i * NHEAD + h] = sr[h]; }
}

// F[N x 256] = X[N x 128] @ Wb^T  (Wb row-major [256 x 128], both bf16).
// Block = 128 threads = 4 waves; block covers 16 rows x 256 cols; each wave
// does 4 col-tiles of 16.  K=128 -> 4 chained v_wmma_f32_16x16x32_bf16.
__global__ __launch_bounds__(128)
void gemm_xw_wmma(const bf16_t* __restrict__ X, const bf16_t* __restrict__ Wb,
                  float* __restrict__ F, int N) {
    const int lane    = threadIdx.x & 31;
    const int wave    = threadIdx.x >> 5;
    const int rowBase = blockIdx.x * 16;
    const int r       = lane & 15;
    const int row     = rowBase + r;
    const int rowc    = (row < N) ? row : (N - 1);  // clamp; stores are guarded
    const int kaOff   = (lane < 16) ? 0 : 16;       // A frag K sub-offset
    const int kbOff   = (lane < 16) ? 0 : 16;       // B frag K sub-offset
    const bool full   = (rowBase + 16 <= N);        // uniform across the block
    const int rhalf   = rowBase + ((lane < 16) ? 0 : 8); // first row this lane stores

    // Preload the 4 A fragments (rows of X) for K blocks 0..3.
    v16bf a[4];
#pragma unroll
    for (int kb = 0; kb < 4; ++kb) {
        const bf16_t* p = X + (long)rowc * IN_F + kb * 32 + kaOff;
        v8bf lo = *(const v8bf*)(p);        // K +0..7
        v8bf hi = *(const v8bf*)(p + 8);    // K +8..15
        a[kb] = __builtin_shufflevector(lo, hi, 0,1,2,3,4,5,6,7,8,9,10,11,12,13,14,15);
    }

#pragma unroll
    for (int t = 0; t < 4; ++t) {
        const int nb  = (wave * 4 + t) * 16;   // output-column tile base
        const int col = nb + r;                // W row == B column (B = W^T)
        v8f c = {0.f, 0.f, 0.f, 0.f, 0.f, 0.f, 0.f, 0.f};
#pragma unroll
        for (int kb = 0; kb < 4; ++kb) {
            const bf16_t* q = Wb + (long)col * IN_F + kb * 32 + kbOff;
            v8bf lo = *(const v8bf*)(q);       // K +0..7  (contiguous in W row)
            v8bf hi = *(const v8bf*)(q + 8);   // K +8..15
            v16bf b = __builtin_shufflevector(lo, hi, 0,1,2,3,4,5,6,7,8,9,10,11,12,13,14,15);
            c = __builtin_amdgcn_wmma_f32_16x16x32_bf16(
                    false, a[kb], false, b, (short)0, c, false, false);
        }
        // D layout: lanes 0-15 -> rows 0..7 (VGPR v = row v), lanes 16-31 -> rows 8..15.
        // One base pointer + immediate row offsets (m*HD*4 = m*1KB) per lane.
        float* outp = F + (long)rhalf * HD + nb + r;
        if (full) {
#pragma unroll
            for (int m = 0; m < 8; ++m) outp[m * HD] = c[m];
        } else {
#pragma unroll
            for (int m = 0; m < 8; ++m)
                if (rhalf + m < N) outp[m * HD] = c[m];
        }
    }
}

// el[n][h] = x[n] . wl[:,h]   (folded attention GEMV, [N x 4] out)
__global__ void node_attn_dot(const float* __restrict__ x, const float* __restrict__ wf,
                              float* __restrict__ out, int N) {
    int n = blockIdx.x * blockDim.x + threadIdx.x;
    if (n >= N) return;
    const float* xr = x + (long)n * IN_F;
    float acc[NHEAD] = {0.f, 0.f, 0.f, 0.f};
    for (int i = 0; i < IN_F; ++i) {
        float xv = xr[i];
        const float* w = wf + i * NHEAD;
#pragma unroll
        for (int h = 0; h < NHEAD; ++h) acc[h] += xv * w[h];
    }
    f32x4 o = {acc[0], acc[1], acc[2], acc[3]};
    *(f32x4*)(out + (long)n * NHEAD) = o;
}

__global__ void init_attn(float* __restrict__ emax, float* __restrict__ den, int n) {
    int i = blockIdx.x * blockDim.x + threadIdx.x;
    if (i < n) { emax[i] = -INFINITY; den[i] = 0.f; }
}

__device__ inline void atomicMaxF(float* addr, float v) {
    if (v >= 0.f) atomicMax((int*)addr, __float_as_int(v));
    else          atomicMin((unsigned int*)addr, __float_as_uint(v));
}

__device__ inline float lrelu(float v) { return v > 0.f ? v : NEG_SLOPE * v; }

__global__ void edge_max(const int* __restrict__ src, const int* __restrict__ dst,
                         const float* __restrict__ el, const float* __restrict__ er,
                         float* __restrict__ emax, int E) {
    int e = blockIdx.x * blockDim.x + threadIdx.x;
    if (e >= E) return;
    int s = src[e], d = dst[e];
    f32x4 l = *(const f32x4*)(el + (long)s * NHEAD);
    f32x4 r = *(const f32x4*)(er + (long)d * NHEAD);
#pragma unroll
    for (int h = 0; h < NHEAD; ++h) atomicMaxF(&emax[(long)d * NHEAD + h], lrelu(l[h] + r[h]));
}

// w = exp(e - emax[dst]); accumulate den and stash w per edge.
__global__ void edge_sum(const int* __restrict__ src, const int* __restrict__ dst,
                         const float* __restrict__ el, const float* __restrict__ er,
                         const float* __restrict__ emax, float* __restrict__ den,
                         float* __restrict__ wbuf, int E) {
    int e = blockIdx.x * blockDim.x + threadIdx.x;
    if (e >= E) return;
    int s = src[e], d = dst[e];
    f32x4 l = *(const f32x4*)(el + (long)s * NHEAD);
    f32x4 r = *(const f32x4*)(er + (long)d * NHEAD);
    f32x4 m = *(const f32x4*)(emax + (long)d * NHEAD);
    f32x4 w;
#pragma unroll
    for (int h = 0; h < NHEAD; ++h) {
        float wv = __expf(lrelu(l[h] + r[h]) - m[h]);
        w[h] = wv;
        atomicAdd(&den[(long)d * NHEAD + h], wv);
    }
    *(f32x4*)(wbuf + (long)e * NHEAD) = w;
}

// Normalize per-edge weights in place: alpha = w / den[dst].
__global__ void edge_alpha(const int* __restrict__ dst, const float* __restrict__ den,
                           float* __restrict__ wbuf, int E) {
    int e = blockIdx.x * blockDim.x + threadIdx.x;
    if (e >= E) return;
    int d = dst[e];
    f32x4 w  = *(const f32x4*)(wbuf + (long)e * NHEAD);
    f32x4 dn = *(const f32x4*)(den + (long)d * NHEAD);
#pragma unroll
    for (int h = 0; h < NHEAD; ++h) w[h] = w[h] / dn[h];
    *(f32x4*)(wbuf + (long)e * NHEAD) = w;
}

// Hot pass: 64 threads per edge (one per output feature); 4 heads each.
// Only loads: alpha4 (16B) + 4 gathered fs dwords; 4 f32 atomics out.
__global__ void edge_acc(const int* __restrict__ src, const int* __restrict__ dst,
                         const float* __restrict__ alpha, const float* __restrict__ F,
                         float* __restrict__ acc, long E) {
    long gid = (long)blockIdx.x * blockDim.x + threadIdx.x;
    long e   = gid >> 6;
    int  dd  = (int)(gid & 63);
    if (e >= E) return;
    int s = src[e], d = dst[e];
    const float* fr = F + (long)s * HD + dd;
    __builtin_prefetch(fr, 0, 0);           // global_prefetch_b8 for gathered row
    f32x4 a4 = *(const f32x4*)(alpha + e * NHEAD);
    float* ao = acc + (long)d * HD + dd;
#pragma unroll
    for (int h = 0; h < NHEAD; ++h)
        atomicAdd(ao + h * DOUT, a4[h] * fr[h * DOUT]);
}

// out[n][d] = 0.25 * ( sum_h acc[n][h][d] + sum_h sum_{rel->type} b[h][d] )
__global__ void finalize(const float* __restrict__ acc, const float* __restrict__ b0,
                         const float* __restrict__ b1, const float* __restrict__ b2,
                         float* __restrict__ out, int N) {
    long gid = (long)blockIdx.x * blockDim.x + threadIdx.x;
    if (gid >= (long)N * DOUT) return;
    long n = gid >> 6;
    int  dd = (int)(gid & 63);
    float s = 0.f;
#pragma unroll
    for (int h = 0; h < NHEAD; ++h) {
        s += acc[n * HD + h * DOUT + dd];
        if (b0) s += b0[h * DOUT + dd];
        if (b1) s += b1[h * DOUT + dd];
        if (b2) s += b2[h * DOUT + dd];
    }
    out[gid] = 0.25f * s;
}

// ------------------------------ host driver --------------------------------

extern "C" void kernel_launch(void* const* d_in, const int* in_sizes, int n_in,
                              void* d_out, int out_size, void* d_ws, size_t ws_size,
                              hipStream_t stream) {
    (void)in_sizes; (void)n_in; (void)out_size; (void)ws_size;

    static const int  NN[3]   = {N_USER, N_ARTIST, N_TAG};
    static const long EE[6]   = {500000, 500000, 250000, 250000, 100000, 100000};
    static const int  srcT[6] = {0, 1, 0, 0, 1, 2};   // ua au uu0 uu1 at ta
    static const int  dstT[6] = {1, 0, 0, 0, 2, 1};

    const float* x[3] = {(const float*)d_in[0], (const float*)d_in[1], (const float*)d_in[2]};

    // -------- workspace carve-out (256B aligned) --------
    char* ws = (char*)d_ws;
    size_t off = 0;
    auto take = [&](size_t bytes) -> char* {
        char* p = ws + off;
        off = (off + bytes + 255) & ~(size_t)255;
        return p;
    };
    bf16_t* xb[3];
    for (int t = 0; t < 3; ++t) xb[t] = (bf16_t*)take((size_t)NN[t] * IN_F * sizeof(bf16_t));
    bf16_t* Wb   = (bf16_t*)take((size_t)HD * IN_F * sizeof(bf16_t));
    float*  wl   = (float*)take((size_t)IN_F * NHEAD * sizeof(float));
    float*  wr   = (float*)take((size_t)IN_F * NHEAD * sizeof(float));
    float*  fs   = (float*)take((size_t)N_USER * HD * sizeof(float));   // max src type
    float*  el   = (float*)take((size_t)N_USER * NHEAD * sizeof(float));
    float*  er   = (float*)take((size_t)N_USER * NHEAD * sizeof(float));
    float*  emax = (float*)take((size_t)N_USER * NHEAD * sizeof(float));
    float*  den  = (float*)take((size_t)N_USER * NHEAD * sizeof(float));
    float*  wbuf = (float*)take((size_t)E_MAX * NHEAD * sizeof(float));
    float*  accb[3];
    for (int t = 0; t < 3; ++t) accb[t] = (float*)take((size_t)NN[t] * HD * sizeof(float));

    // -------- one-time per call: bf16 features + zeroed accumulators --------
    for (int t = 0; t < 3; ++t) {
        long n = (long)NN[t] * IN_F;
        cvt_f32_to_bf16<<<cdiv(n, 256), 256, 0, stream>>>(x[t], xb[t], n);
        hipMemsetAsync(accb[t], 0, (size_t)NN[t] * HD * sizeof(float), stream);
    }

    // -------- per relation --------
    for (int rix = 0; rix < 6; ++rix) {
        const int s = srcT[rix], d = dstT[rix];
        const float* W   = (const float*)d_in[3 + 4 * rix + 0];
        const float* al  = (const float*)d_in[3 + 4 * rix + 1];
        const float* arp = (const float*)d_in[3 + 4 * rix + 2];
        const int*   es  = (const int*)d_in[27 + 2 * rix + 0];
        const int*   ed  = (const int*)d_in[27 + 2 * rix + 1];
        const long   E   = EE[rix];

        cvt_f32_to_bf16<<<cdiv((long)HD * IN_F, 256), 256, 0, stream>>>(W, Wb, (long)HD * IN_F);
        fold_attn<<<1, 128, 0, stream>>>(W, al, arp, wl, wr);

        gemm_xw_wmma<<<cdiv(NN[s], 16), 128, 0, stream>>>(xb[s], Wb, fs, NN[s]);

        node_attn_dot<<<cdiv(NN[s], 256), 256, 0, stream>>>(x[s], wl, el, NN[s]);
        node_attn_dot<<<cdiv(NN[d], 256), 256, 0, stream>>>(x[d], wr, er, NN[d]);

        init_attn<<<cdiv((long)NN[d] * NHEAD, 256), 256, 0, stream>>>(emax, den, NN[d] * NHEAD);
        edge_max<<<cdiv(E, 256), 256, 0, stream>>>(es, ed, el, er, emax, (int)E);
        edge_sum<<<cdiv(E, 256), 256, 0, stream>>>(es, ed, el, er, emax, den, wbuf, (int)E);
        edge_alpha<<<cdiv(E, 256), 256, 0, stream>>>(ed, den, wbuf, (int)E);
        edge_acc<<<cdiv(E * 64, 256), 256, 0, stream>>>(es, ed, wbuf, fs, accb[d], E);
    }

    // -------- finalize: head mean + per-relation biases, per dst type --------
    float* out = (float*)d_out;
    const float* bAU  = (const float*)d_in[3 + 4 * 1 + 3];   // au  -> user
    const float* bUU0 = (const float*)d_in[3 + 4 * 2 + 3];   // uu0 -> user
    const float* bUU1 = (const float*)d_in[3 + 4 * 3 + 3];   // uu1 -> user
    const float* bUA  = (const float*)d_in[3 + 4 * 0 + 3];   // ua  -> artist
    const float* bTA  = (const float*)d_in[3 + 4 * 5 + 3];   // ta  -> artist
    const float* bAT  = (const float*)d_in[3 + 4 * 4 + 3];   // at  -> tag

    finalize<<<cdiv((long)N_USER * DOUT, 256), 256, 0, stream>>>(
        accb[0], bAU, bUU0, bUU1, out, N_USER);
    finalize<<<cdiv((long)N_ARTIST * DOUT, 256), 256, 0, stream>>>(
        accb[1], bUA, bTA, nullptr, out + (long)N_USER * DOUT, N_ARTIST);
    finalize<<<cdiv((long)N_TAG * DOUT, 256), 256, 0, stream>>>(
        accb[2], bAT, nullptr, nullptr, out + (long)(N_USER + N_ARTIST) * DOUT, N_TAG);
}
